// RNN_44109314130098
// MI455X (gfx1250) — matrix operand order repaired and checked by reading, not compiled, not run on hardware
//
#include <hip/hip_runtime.h>

typedef __attribute__((ext_vector_type(16))) _Float16 v16h;
typedef __attribute__((ext_vector_type(8)))  _Float16 v8h;
typedef __attribute__((ext_vector_type(2)))  _Float16 v2h;
typedef __attribute__((ext_vector_type(8)))  float    v8f;

#define SEQ 2048
#define DIN 50
#define HID 64
#define WST 72   // LDS row stride (f16): 144B -> 16B aligned, conflict-free 16B fragment loads

union V16 { v16h v; v8h h[2]; };

__device__ __forceinline__ v8f wmma16(v16h a, v16h b, v8f c) {
  // D = A(16x32 f16) * B(32x16 f16) + C(16x16 f32)
  return __builtin_amdgcn_wmma_f32_16x16x32_f16(false, a, false, b, (short)0, c, false, false);
}

// Prefer native v_tanh_f32 (CDNA5 TRANS op); fall back to exp2+rcp sequence.
__device__ __forceinline__ float tanh_fast(float x) {
#if __has_builtin(__builtin_amdgcn_tanhf)
  return __builtin_amdgcn_tanhf(x);
#else
  // tanh(x) = 1 - 2/(exp2(x * 2*log2 e) + 1)
  float e = __builtin_amdgcn_exp2f(x * 2.885390081777927f);
  return 1.0f - 2.0f * __builtin_amdgcn_rcpf(e + 1.0f);
#endif
}

// CDNA5 split workgroup barrier: drain LDS then signal; wait separately so
// independent work (next-step x staging) can sit between signal and wait.
__device__ __forceinline__ void bar_arrive() {
  __asm volatile("s_wait_dscnt 0\n\ts_barrier_signal -1" ::: "memory");
}
__device__ __forceinline__ void bar_wait() {
  __asm volatile("s_barrier_wait -1" ::: "memory");
}

__global__ __launch_bounds__(128) void rnn2_head_kernel(
    const float* __restrict__ x,
    const float* __restrict__ Wih0, const float* __restrict__ Whh0,
    const float* __restrict__ bih0, const float* __restrict__ bhh0,
    const float* __restrict__ Wih1, const float* __restrict__ Whh1,
    const float* __restrict__ bih1, const float* __restrict__ bhh1,
    const float* __restrict__ W1,   const float* __restrict__ b1,
    const float* __restrict__ W2,   const float* __restrict__ b2,
    float* __restrict__ out)
{
  __shared__ __align__(16) _Float16 wih0[HID*WST];
  __shared__ __align__(16) _Float16 whh0[HID*WST];
  __shared__ __align__(16) _Float16 wih1[HID*WST];
  __shared__ __align__(16) _Float16 whh1[HID*WST];
  __shared__ __align__(16) _Float16 xsA[16*WST];   // double-buffered x stage
  __shared__ __align__(16) _Float16 xsB[16*WST];
  __shared__ __align__(16) _Float16 h1s[16*WST];
  __shared__ __align__(16) _Float16 h2s[16*WST];

  const int tid  = threadIdx.x;
  const int lane = tid & 31;
  const int nt   = tid >> 5;       // wave id == N-tile (16 hidden cols)
  const int b0   = blockIdx.x * 16;
  const int col  = lane & 15;
  const int kh   = lane >> 4;

  // ---- one-time: weights f32 -> f16 LDS (Wih0 zero-padded 50->64) ----
  for (int i = tid; i < HID*HID; i += 128) {
    int r = i >> 6, c = i & 63;
    wih0[r*WST + c] = (c < DIN) ? (_Float16)Wih0[r*DIN + c] : (_Float16)0.0f;
    whh0[r*WST + c] = (_Float16)Whh0[i];
    wih1[r*WST + c] = (_Float16)Wih1[i];
    whh1[r*WST + c] = (_Float16)Whh1[i];
  }
  for (int i = tid; i < 16*WST; i += 128) {
    xsA[i] = (_Float16)0.0f; xsB[i] = (_Float16)0.0f;
    h1s[i] = (_Float16)0.0f; h2s[i] = (_Float16)0.0f;
  }
  __syncthreads();

  // ---- hoist B-fragments (B[k][n] = W[n][k]) into registers ----
  const int wrow = nt*16 + col;
  V16 fih0[2], fhh0[2], fih1[2], fhh1[2];
#pragma unroll
  for (int kt = 0; kt < 2; ++kt) {
    const int o = wrow*WST + kt*32 + kh*8;
    fih0[kt].h[0] = *(const v8h*)&wih0[o]; fih0[kt].h[1] = *(const v8h*)&wih0[o+16];
    fhh0[kt].h[0] = *(const v8h*)&whh0[o]; fhh0[kt].h[1] = *(const v8h*)&whh0[o+16];
    fih1[kt].h[0] = *(const v8h*)&wih1[o]; fih1[kt].h[1] = *(const v8h*)&wih1[o+16];
    fhh1[kt].h[0] = *(const v8h*)&whh1[o]; fhh1[kt].h[1] = *(const v8h*)&whh1[o+16];
  }
  const float bias0 = bih0[wrow] + bhh0[wrow];
  const float bias1 = bih1[wrow] + bhh1[wrow];

  V16 h1f[2] = {};                        // loop-carried h1 A-fragments
  const int ab   = col*WST + kh*8;        // A-fragment base (lane layout)
  const int xrow = tid >> 3;              // 8 threads per x row
  const int c2   = tid & 7;               // float2 slot within row
  const float* xbase = x + ((long)(b0 + xrow) * SEQ) * DIN;

  // branchless vectorized staging: 16 rows x 50 f32 -> f16 (cols 50..63 stay 0)
  auto stageX = [&](int s, _Float16* dst) {
    const float2* xr2 = (const float2*)(xbase + (long)s * DIN);
#pragma unroll
    for (int j = 0; j < 3; ++j) {
      float2 t = xr2[c2 + 8*j];
      v2h p; p.x = (_Float16)t.x; p.y = (_Float16)t.y;
      *(v2h*)&dst[xrow*WST + (c2 + 8*j)*2] = p;
    }
    if (c2 == 0) {                        // cols 48,49
      float2 t = xr2[24];
      v2h p; p.x = (_Float16)t.x; p.y = (_Float16)t.y;
      *(v2h*)&dst[xrow*WST + 48] = p;
    }
  };

  auto step = [&](int s, const _Float16* xsR, _Float16* xsW) {
    bar_wait();                            // prev-step h2 + this-step x visible
    V16 xf0, xf1, h2f0, h2f1;
    xf0.h[0]  = *(const v8h*)&xsR[ab];     xf0.h[1]  = *(const v8h*)&xsR[ab+16];
    xf1.h[0]  = *(const v8h*)&xsR[ab+32];  xf1.h[1]  = *(const v8h*)&xsR[ab+48];
    h2f0.h[0] = *(const v8h*)&h2s[ab];     h2f0.h[1] = *(const v8h*)&h2s[ab+16];
    h2f1.h[0] = *(const v8h*)&h2s[ab+32];  h2f1.h[1] = *(const v8h*)&h2s[ab+48];

    v8f acc;                               // layer 0
#pragma unroll
    for (int i = 0; i < 8; ++i) acc[i] = bias0;
    acc = wmma16(xf0.v,    fih0[0].v, acc);
    acc = wmma16(xf1.v,    fih0[1].v, acc);
    acc = wmma16(h1f[0].v, fhh0[0].v, acc);
    acc = wmma16(h1f[1].v, fhh0[1].v, acc);
#pragma unroll
    for (int i = 0; i < 8; ++i)
      h1s[(i + 8*kh)*WST + nt*16 + col] = (_Float16)tanh_fast(acc[i]);
    bar_arrive();                          // h1 handoff (signal only)

    stageX((s + 1 < SEQ) ? s + 1 : SEQ - 1, xsW);  // overlap with other waves' L0

    bar_wait();                            // h1 complete everywhere
    h1f[0].h[0] = *(const v8h*)&h1s[ab];    h1f[0].h[1] = *(const v8h*)&h1s[ab+16];
    h1f[1].h[0] = *(const v8h*)&h1s[ab+32]; h1f[1].h[1] = *(const v8h*)&h1s[ab+48];

    v8f acc2;                              // layer 1
#pragma unroll
    for (int i = 0; i < 8; ++i) acc2[i] = bias1;
    acc2 = wmma16(h1f[0].v, fih1[0].v, acc2);
    acc2 = wmma16(h1f[1].v, fih1[1].v, acc2);
    acc2 = wmma16(h2f0.v,   fhh1[0].v, acc2);
    acc2 = wmma16(h2f1.v,   fhh1[1].v, acc2);
#pragma unroll
    for (int i = 0; i < 8; ++i)
      h2s[(i + 8*kh)*WST + nt*16 + col] = (_Float16)tanh_fast(acc2[i]);
    bar_arrive();                          // end-of-step handoff
  };

  stageX(0, xsA);
  bar_arrive();
  for (int s = 0; s < SEQ; s += 2) {       // unroll-by-2: static buffer parity
    step(s,     xsA, xsB);
    step(s + 1, xsB, xsA);
  }
  bar_wait();                              // final h2 visible

  // ---- head: sigmoid(relu(h2 @ W1^T + b1) @ W2^T + b2) ----
  if (tid < 16) {
    float a2 = b2[0];
    for (int j = 0; j < 32; ++j) {
      float hj = b1[j];
#pragma unroll 8
      for (int k = 0; k < 64; ++k)
        hj += (float)h2s[tid*WST + k] * W1[j*64 + k];
      a2 += fmaxf(hj, 0.0f) * W2[j];
    }
    out[b0 + tid] = __builtin_amdgcn_rcpf(1.0f + __expf(-a2));
  }
}

extern "C" void kernel_launch(void* const* d_in, const int* in_sizes, int n_in,
                              void* d_out, int out_size, void* d_ws, size_t ws_size,
                              hipStream_t stream) {
  (void)in_sizes; (void)n_in; (void)out_size; (void)d_ws; (void)ws_size;
  const float* x    = (const float*)d_in[0];
  const float* Wih0 = (const float*)d_in[1];
  const float* Whh0 = (const float*)d_in[2];
  const float* bi0  = (const float*)d_in[3];
  const float* bh0  = (const float*)d_in[4];
  const float* Wih1 = (const float*)d_in[5];
  const float* Whh1 = (const float*)d_in[6];
  const float* bi1  = (const float*)d_in[7];
  const float* bh1  = (const float*)d_in[8];
  const float* W1   = (const float*)d_in[9];
  const float* b1   = (const float*)d_in[10];
  const float* W2   = (const float*)d_in[11];
  const float* b2   = (const float*)d_in[12];
  rnn2_head_kernel<<<dim3(512/16), dim3(128), 0, stream>>>(
      x, Wih0, Whh0, bi0, bh0, Wih1, Whh1, bi1, bh1, W1, b1, W2, b2, (float*)d_out);
}
